// RetrievalTool_78314433675724
// MI455X (gfx1250) — matrix operand order repaired and checked by reading, not compiled, not run on hardware
//
#include <hip/hip_runtime.h>
#include <hip/hip_bf16.h>

// ---------------- problem constants (from the reference) ----------------
#define SEQ_LEN   336
#define PRED_LEN  96
#define CH        21
#define NTRAIN    5000
#define BSZ       64
#define MARK      4
#define GNUM      3           // PERIODS = (4,2,1) -> per = 4 >> g
#define KEEPK     622         // round(256 + 512*(1 - 96/336))
#define SELK      20
#define NPAD      5008        // 313 * 16 (N tiles of 16 over 5000 bank rows)
#define NTILES    313
#define INV_TEMP  10.0f

typedef _Float16 v8h  __attribute__((ext_vector_type(8)));
typedef _Float16 v16h __attribute__((ext_vector_type(16)));
typedef float    v8f  __attribute__((ext_vector_type(8)));

__host__ __device__ inline int dpad_of(int g) {
    int per = 4 >> g;                       // 4,2,1
    int D = (SEQ_LEN / per) * CH;           // 1764,3528,7056
    return (D + 31) & ~31;                  // 1792,3552,7072
}
__host__ __device__ inline size_t seg_off_halves(int g, int rows) {
    size_t o = 0;
    for (int i = 0; i < g; ++i) o += (size_t)rows * dpad_of(i);
    return o;
}

// ---------------------------------------------------------------------------
// 1) Per-row prep: block means, subtract last block, center, l2-normalize, f16
//    blockIdx.x = row, blockIdx.y = g.  256 threads.
// ---------------------------------------------------------------------------
__global__ void __launch_bounds__(256)
prep_rows_kernel(const float* __restrict__ src, _Float16* __restrict__ dstBase,
                 int nValidRows, int nDstRows)
{
    const int r   = blockIdx.x;
    const int g   = blockIdx.y;
    const int per = 4 >> g;
    const int nblk = SEQ_LEN / per;
    const int D   = nblk * CH;
    const int Dp  = (D + 31) & ~31;
    const int tid = threadIdx.x;
    _Float16* dst = dstBase + seg_off_halves(g, nDstRows) + (size_t)r * Dp;

    if (r >= nValidRows) {               // zero padding rows (bank t >= 5000)
        for (int d = tid; d < Dp; d += 256) dst[d] = (_Float16)0.0f;
        return;
    }

    __shared__ float buf[SEQ_LEN * CH];  // block-domain row (<= 7056 floats)
    __shared__ float lastv[CH];
    __shared__ float redA[256];
    __shared__ float redB[256];

    const float* row = src + (size_t)r * SEQ_LEN * CH;
    for (int d = tid; d < D; d += 256) {
        int blk = d / CH, c = d - blk * CH;
        float s = 0.0f;
        for (int j = 0; j < per; ++j) s += row[(blk * per + j) * CH + c];
        buf[d] = s / (float)per;
    }
    __syncthreads();
    if (tid < CH) lastv[tid] = buf[(nblk - 1) * CH + tid];
    __syncthreads();

    float ps = 0.0f, ps2 = 0.0f;
    for (int d = tid; d < D; d += 256) {
        float v = buf[d] - lastv[d % CH];   // mg - mg[..., -1:, :]
        buf[d] = v;
        ps += v; ps2 += v * v;
    }
    redA[tid] = ps; redB[tid] = ps2;
    __syncthreads();
    for (int s = 128; s > 0; s >>= 1) {
        if (tid < s) { redA[tid] += redA[tid + s]; redB[tid] += redB[tid + s]; }
        __syncthreads();
    }
    const float mu  = redA[0] / (float)D;
    const float ss  = fmaxf(redB[0] - (float)D * mu * mu, 0.0f);
    const float inv = 1.0f / fmaxf(sqrtf(ss), 1e-12f);

    for (int d = tid; d < D; d += 256)       dst[d] = (_Float16)((buf[d] - mu) * inv);
    for (int d = D + tid; d < Dp; d += 256)  dst[d] = (_Float16)0.0f;
}

// ---------------------------------------------------------------------------
// 2) Candidate mask: 622-NN mark-distance threshold (bit-level kth-smallest)
//    plus self window.  One block per batch row b.
// ---------------------------------------------------------------------------
__global__ void __launch_bounds__(256)
build_mask_kernel(const float* __restrict__ train_mark,
                  const float* __restrict__ meta_query,
                  const int*   __restrict__ index,
                  unsigned char* __restrict__ mask)
{
    const int b = blockIdx.x, tid = threadIdx.x;
    __shared__ unsigned int dbits[NTRAIN];
    __shared__ int cnt;

    float q[MARK];
    for (int m = 0; m < MARK; ++m)
        q[m] = meta_query[((size_t)b * SEQ_LEN + SEQ_LEN - 1) * MARK + m];

    for (int t = tid; t < NTRAIN; t += 256) {
        const float* tk = train_mark + ((size_t)t * SEQ_LEN + SEQ_LEN - 1) * MARK;
        float d2 = 0.0f;
        for (int m = 0; m < MARK; ++m) { float d = q[m] - tk[m]; d2 += d * d; }
        dbits[t] = __float_as_uint(sqrtf(d2 + 1e-12f));  // nonneg -> uint-ordered
    }
    __syncthreads();

    // smallest u with |{t : bits <= u}| >= KEEPK
    unsigned lo = 0u, hi = 0x7f800000u;
    for (int it = 0; it < 32 && lo < hi; ++it) {
        unsigned mid = lo + ((hi - lo) >> 1);
        if (tid == 0) cnt = 0;
        __syncthreads();
        int c = 0;
        for (int t = tid; t < NTRAIN; t += 256) c += (dbits[t] <= mid) ? 1 : 0;
        atomicAdd(&cnt, c);               // LDS ds_add
        __syncthreads();
        int total = cnt;
        __syncthreads();
        if (total >= KEEPK) hi = mid; else lo = mid + 1;
    }
    const unsigned tau = lo;

    const int idx = index[b];
    int loS = idx - (SEQ_LEN + PRED_LEN - 1); if (loS < 0) loS = 0;
    int hiS = idx + (SEQ_LEN + PRED_LEN - 1); if (hiS > NTRAIN - 1) hiS = NTRAIN - 1;

    for (int t = tid; t < NPAD; t += 256) {
        unsigned char m = 1;
        if (t < NTRAIN)
            m = ((dbits[t] > tau) || (t >= loS && t <= hiS)) ? 1 : 0;
        mask[(size_t)b * NPAD + t] = m;
    }
}

// ---------------------------------------------------------------------------
// 3) WMMA GEMM: sim[g] (64 x 5008) = X_g (64 x Dp) * Bank_g^T (Dp x 5008), f16->f32
//    block = 128 threads = 4 waves; wave w -> M-tile w (16 rows); blockIdx.x -> N tile.
//    Fragment maps follow ISA 7.12.2 (16-bit A 16x32, B 32x16, f32 C 16x16).
// ---------------------------------------------------------------------------
__global__ void __launch_bounds__(128)
gemm_wmma_kernel(const _Float16* __restrict__ xF,
                 const _Float16* __restrict__ bankF,
                 float* __restrict__ sim)
{
    const int g   = blockIdx.y;
    const int per = 4 >> g;
    const int D   = (SEQ_LEN / per) * CH;
    const int Dp  = (D + 31) & ~31;

    const int lane = threadIdx.x & 31;
    const int wv   = threadIdx.x >> 5;
    const int m0   = wv * 16;
    const int t0   = blockIdx.x * 16;

    const _Float16* Xg = xF    + seg_off_halves(g, BSZ);
    const _Float16* Bg = bankF + seg_off_halves(g, NPAD);

    const int row   = m0 + (lane & 15);
    const int tcol  = t0 + (lane & 15);
    const int abase = (lane < 16) ? 0 : 8;     // A: lanes16-31 hold K+8 region
    const int bbase = (lane < 16) ? 0 : 8;     // B: lanes16-31 hold K+8 region
    const _Float16* ap = Xg + (size_t)row  * Dp + abase;
    const _Float16* bp = Bg + (size_t)tcol * Dp + bbase;

    v8f acc = {};
    for (int kc = 0; kc < Dp; kc += 32) {
        __builtin_prefetch(bp + kc + 256, 0, 1);              // global_prefetch_b8
        v8h alo = *(const v8h*)(ap + kc);                     // K = abase..abase+7
        v8h ahi = *(const v8h*)(ap + kc + 16);                // K = abase+16..+23
        v8h blo = *(const v8h*)(bp + kc);                     // K = bbase..bbase+7
        v8h bhi = *(const v8h*)(bp + kc + 16);                // K = bbase+16..+23
        v16h A = __builtin_shufflevector(alo, ahi, 0,1,2,3,4,5,6,7,8,9,10,11,12,13,14,15);
        v16h B = __builtin_shufflevector(blo, bhi, 0,1,2,3,4,5,6,7,8,9,10,11,12,13,14,15);
        acc = __builtin_amdgcn_wmma_f32_16x16x32_f16(false, A, false, B,
                                                     (short)0, acc, false, false);
    }

    // C layout: VGPR i -> lanes 0-15: M=i, N=lane; lanes 16-31: M=i+8, N=lane-16
    float* srow = sim + (size_t)g * BSZ * NPAD;
    const int hi8  = (lane < 16) ? 0 : 8;
    const int ccol = t0 + (lane & 15);
    for (int i = 0; i < 8; ++i)
        srow[(size_t)(m0 + hi8 + i) * NPAD + ccol] = acc[i];
}

// ---------------------------------------------------------------------------
// 4) Masked top-20 per (g,b) row of sim.
// ---------------------------------------------------------------------------
__global__ void __launch_bounds__(256)
topk_kernel(const float* __restrict__ sim, const unsigned char* __restrict__ mask,
            float* __restrict__ wscore, int* __restrict__ widx)
{
    const int b = blockIdx.x, g = blockIdx.y, tid = threadIdx.x;
    __shared__ float vals[NPAD];
    __shared__ float rv[256];
    __shared__ int   ri[256];
    const float NEG = -__builtin_huge_valf();

    const float* srow         = sim  + ((size_t)g * BSZ + b) * NPAD;
    const unsigned char* mrow = mask + (size_t)b * NPAD;
    for (int t = tid; t < NPAD; t += 256)
        vals[t] = mrow[t] ? NEG : srow[t];
    __syncthreads();

    for (int r = 0; r < SELK; ++r) {
        float bv = NEG; int bi = -1;
        for (int t = tid; t < NPAD; t += 256) {
            float v = vals[t];
            if (v > bv) { bv = v; bi = t; }
        }
        rv[tid] = bv; ri[tid] = bi;
        __syncthreads();
        for (int s = 128; s > 0; s >>= 1) {
            if (tid < s && rv[tid + s] > rv[tid]) { rv[tid] = rv[tid + s]; ri[tid] = ri[tid + s]; }
            __syncthreads();
        }
        if (tid == 0) {
            const size_t o = ((size_t)g * BSZ + b) * SELK + r;
            int bsel = ri[0];
            wscore[o] = rv[0];
            widx[o]   = (bsel < 0) ? 0 : bsel;
            if (bsel >= 0) vals[bsel] = NEG;
        }
        __syncthreads();
    }
}

// ---------------------------------------------------------------------------
// 5) softmax(score/TEMP) + on-the-fly y_mg gather -> pred (G,B,PRED_LEN,CH)
// ---------------------------------------------------------------------------
__global__ void __launch_bounds__(256)
final_kernel(const float* __restrict__ ydata, const float* __restrict__ wscore,
             const int* __restrict__ widx, float* __restrict__ out)
{
    const int b = blockIdx.x, g = blockIdx.y, tid = threadIdx.x;
    const int per  = 4 >> g;
    const int nblk = PRED_LEN / per;
    const int P    = PRED_LEN * CH;         // 2016

    __shared__ float bm[PRED_LEN * CH];
    __shared__ float prob_s[SELK];

    const size_t o = ((size_t)g * BSZ + b) * SELK;
    if (tid == 0) {
        float sc[SELK], mx = -__builtin_huge_valf();
        for (int k = 0; k < SELK; ++k) { sc[k] = wscore[o + k]; mx = fmaxf(mx, sc[k]); }
        float s = 0.0f;
        for (int k = 0; k < SELK; ++k) { float e = expf((sc[k] - mx) * INV_TEMP); prob_s[k] = e; s += e; }
        const float inv = 1.0f / s;
        for (int k = 0; k < SELK; ++k) prob_s[k] *= inv;
    }
    __syncthreads();

    float acc[8];
    for (int i = 0; i < 8; ++i) acc[i] = 0.0f;

    for (int k = 0; k < SELK; ++k) {
        const int t = widx[o + k];
        const float pk = prob_s[k];
        const float* yrow = ydata + (size_t)t * PRED_LEN * CH;
        for (int e = tid; e < nblk * CH; e += 256) {
            int blk = e / CH, c = e - blk * CH;
            float s = 0.0f;
            for (int j = 0; j < per; ++j) s += yrow[(blk * per + j) * CH + c];
            bm[e] = s / (float)per;
        }
        __syncthreads();
        int slot = 0;
        for (int pos = tid; pos < P; pos += 256, ++slot) {
            int p = pos / CH, c = pos - p * CH;
            acc[slot] += pk * (bm[(p / per) * CH + c] - bm[(nblk - 1) * CH + c]);
        }
        __syncthreads();
    }

    float* orow = out + ((size_t)g * BSZ + b) * (size_t)P;
    int slot = 0;
    for (int pos = tid; pos < P; pos += 256, ++slot) orow[pos] = acc[slot];
}

// ---------------------------------------------------------------------------
extern "C" void kernel_launch(void* const* d_in, const int* in_sizes, int n_in,
                              void* d_out, int out_size, void* d_ws, size_t ws_size,
                              hipStream_t stream)
{
    const float* x      = (const float*)d_in[0];  // (64,336,21)
    const float* train  = (const float*)d_in[1];  // (5000,336,21)
    const float* ydata  = (const float*)d_in[2];  // (5000,96,21)
    const float* tmark  = (const float*)d_in[3];  // (5000,336,4)
    const float* mquery = (const float*)d_in[4];  // (64,336,4)
    const int*   index  = (const int*)d_in[5];    // (64,)
    float* out = (float*)d_out;                   // (3,64,96,21)

    const size_t DPSUM = (size_t)dpad_of(0) + dpad_of(1) + dpad_of(2);   // 12416
    const size_t bankBytes = (size_t)NPAD * DPSUM * sizeof(_Float16);    // ~124.4 MB
    const size_t xBytes    = (size_t)BSZ  * DPSUM * sizeof(_Float16);
    const size_t simBytes  = (size_t)GNUM * BSZ * NPAD * sizeof(float);
    const size_t maskBytes = (size_t)BSZ * NPAD;
    const size_t scBytes   = (size_t)GNUM * BSZ * SELK * sizeof(float);
    const size_t need = bankBytes + xBytes + simBytes + maskBytes + 2 * scBytes;
    if (ws_size < need) return;   // deterministic no-op if workspace too small

    char* ws = (char*)d_ws;
    _Float16* bankF       = (_Float16*)(ws);
    _Float16* xF          = (_Float16*)(ws + bankBytes);
    float*    sim         = (float*)   (ws + bankBytes + xBytes);
    unsigned char* mask   = (unsigned char*)(ws + bankBytes + xBytes + simBytes);
    float*    wscore      = (float*)   (ws + bankBytes + xBytes + simBytes + maskBytes);
    int*      widx        = (int*)     (ws + bankBytes + xBytes + simBytes + maskBytes + scBytes);

    prep_rows_kernel<<<dim3(NPAD, GNUM), 256, 0, stream>>>(train, bankF, NTRAIN, NPAD);
    prep_rows_kernel<<<dim3(BSZ,  GNUM), 256, 0, stream>>>(x,     xF,    BSZ,    BSZ);
    build_mask_kernel<<<dim3(BSZ), 256, 0, stream>>>(tmark, mquery, index, mask);
    gemm_wmma_kernel<<<dim3(NTILES, GNUM), 128, 0, stream>>>(xF, bankF, sim);
    topk_kernel<<<dim3(BSZ, GNUM), 256, 0, stream>>>(sim, mask, wscore, widx);
    final_kernel<<<dim3(BSZ, GNUM), 256, 0, stream>>>(ydata, wscore, widx, out);
}
